// AIFI_Flash_61435212202332
// MI455X (gfx1250) — compile-verified, matrix operand order checked by reading, not run on hardware
//
#include <hip/hip_runtime.h>
#include <cstdint>
#include <cstddef>

// ---------------- CDNA5 WMMA types ----------------
typedef __attribute__((ext_vector_type(16))) __bf16 v16bf;
typedef __attribute__((ext_vector_type(8)))  float  v8f;
typedef __attribute__((ext_vector_type(4)))  int    v4i;

union Frag16 {
  v16bf v;
  uint4 q[2];
};

// A/B fragment layout for V_WMMA_F32_16X16X32_BF16 (ISA 7.12.2):
// lane L: row/col = L%16, half hh = L>>4.
// elements 0..7  hold K = kb + hh*8     .. +7   (16 contiguous bytes)
// elements 8..15 hold K = kb + 16+hh*8  .. +7   (16 contiguous bytes)
__device__ __forceinline__ void load_frag(const __bf16* row, int kb, int hh, Frag16& f) {
  f.q[0] = *(const uint4*)(row + kb + hh * 8);
  f.q[1] = *(const uint4*)(row + kb + 16 + hh * 8);
}
__device__ __forceinline__ void zero_frag(Frag16& f) {
  f.q[0] = make_uint4(0u, 0u, 0u, 0u);
  f.q[1] = make_uint4(0u, 0u, 0u, 0u);
}
__device__ __forceinline__ v8f wmma_bf16(const Frag16& a, const Frag16& b, v8f c) {
  return __builtin_amdgcn_wmma_f32_16x16x32_bf16(false, a.v, false, b.v, (short)0, c, false, false);
}

// ---------------- async global->LDS staging (gfx1250) ----------------
#if defined(__HIP_DEVICE_COMPILE__) && __has_builtin(__builtin_amdgcn_global_load_async_to_lds_b128)
#define USE_ASYNC 1
#else
#define USE_ASYNC 0
#endif

#define AS1 __attribute__((address_space(1)))
#define AS3 __attribute__((address_space(3)))

__device__ __forceinline__ void copy_b128(const __bf16* g, __bf16* l) {
#if USE_ASYNC
  __builtin_amdgcn_global_load_async_to_lds_b128((AS1 v4i*)g, (AS3 v4i*)l, 0, 0);
#else
  *(uint4*)l = *(const uint4*)g;
#endif
}
__device__ __forceinline__ void wait_async_all() {
#if USE_ASYNC
#if __has_builtin(__builtin_amdgcn_s_wait_asynccnt)
  __builtin_amdgcn_s_wait_asynccnt(0);
#else
  asm volatile("s_wait_asynccnt 0x0" ::: "memory");
#endif
#endif
}

// ---------------- problem constants ----------------
#define BSZ   256
#define CCH   256
#define NTOK  196
#define BN    50176      // BSZ*NTOK
#define CM    2048
#define HEADS 8
#define HD    32
#define EPSBN 1e-5f
#define KC    128        // K-chunk staged in LDS
#define PANEL (64 * KC)  // elements per LDS chunk buffer

// ---------------- block-level GEMM core ----------------
// Block: 256 threads (8 waves). Block computes 256 rows x 64 cols of C.
// Wave w handles rows [mblk*256 + w*32, +32). All waves share the Bt panel,
// staged in LDS in double-buffered 64x128 chunks (async-to-LDS on gfx1250).
// A: [M][K] bf16 row-major; Bt: [N][K] bf16 row-major (B column-major).
__device__ __forceinline__ void stage_chunk(const __bf16* __restrict__ bpan, int K,
                                            int kc, __bf16* buf, int tid) {
#pragma unroll
  for (int i = 0; i < 4; ++i) {
    int idx = i * 256 + tid;          // 1024 x b128 transfers = 64 rows x 128 cols
    int row = idx >> 4;
    int seg = idx & 15;
    copy_b128(bpan + (size_t)row * K + kc + seg * 8, buf + row * KC + seg * 8);
  }
}

__device__ __forceinline__ void gemm_block_acc(const __bf16* __restrict__ A,
                                               const __bf16* __restrict__ Bt,
                                               int K, int mbase, int nbase,
                                               __bf16* Bs, v8f acc[2][4]) {
  const int tid  = threadIdx.x;
  const int lane = tid & 31;
  const int rl = lane & 15, hh = lane >> 4;
  const __bf16* ar0 = A + (size_t)(mbase + rl) * K;
  const __bf16* ar1 = A + (size_t)(mbase + 16 + rl) * K;
  const __bf16* bpan = Bt + (size_t)nbase * K;
  const int nch = K / KC;

  stage_chunk(bpan, K, 0, Bs, tid);
  wait_async_all();
  __syncthreads();

  for (int c = 0; c < nch; ++c) {
    __bf16* cur = Bs + (c & 1) * PANEL;
    if (c + 1 < nch)
      stage_chunk(bpan, K, (c + 1) * KC, Bs + ((c + 1) & 1) * PANEL, tid);
    const __bf16* b0 = cur + rl * KC;
    const __bf16* b1 = cur + (16 + rl) * KC;
    const __bf16* b2 = cur + (32 + rl) * KC;
    const __bf16* b3 = cur + (48 + rl) * KC;
    const int kc = c * KC;
#pragma unroll
    for (int kb = 0; kb < KC; kb += 32) {
      Frag16 a0, a1, f0, f1, f2, f3;
      load_frag(ar0, kc + kb, hh, a0);
      load_frag(ar1, kc + kb, hh, a1);
      load_frag(b0, kb, hh, f0);
      load_frag(b1, kb, hh, f1);
      load_frag(b2, kb, hh, f2);
      load_frag(b3, kb, hh, f3);
      acc[0][0] = wmma_bf16(a0, f0, acc[0][0]);
      acc[0][1] = wmma_bf16(a0, f1, acc[0][1]);
      acc[0][2] = wmma_bf16(a0, f2, acc[0][2]);
      acc[0][3] = wmma_bf16(a0, f3, acc[0][3]);
      acc[1][0] = wmma_bf16(a1, f0, acc[1][0]);
      acc[1][1] = wmma_bf16(a1, f1, acc[1][1]);
      acc[1][2] = wmma_bf16(a1, f2, acc[1][2]);
      acc[1][3] = wmma_bf16(a1, f3, acc[1][3]);
    }
    __syncthreads();                       // all waves done reading `cur`
    if (c + 1 < nch) {
      wait_async_all();                    // next chunk landed in LDS
      __syncthreads();
    }
  }
}

#define GEMM_PROLOGUE(KDIM, NCOLS)                                            \
  __shared__ __bf16 Bs[2 * PANEL];                                            \
  const int lane = threadIdx.x & 31;                                          \
  const int wv   = threadIdx.x >> 5;                                          \
  const int Nt   = (NCOLS) / 64;                                              \
  const int mblk = blockIdx.x / Nt, nblk = blockIdx.x % Nt;                   \
  const int mbase = mblk * 256 + wv * 32;                                     \
  const int nbase = nblk * 64;                                                \
  const int rl = lane & 15, hh = lane >> 4;                                   \
  v8f zero = {0.f, 0.f, 0.f, 0.f, 0.f, 0.f, 0.f, 0.f};                        \
  v8f acc[2][4] = {{zero, zero, zero, zero}, {zero, zero, zero, zero}};       \
  gemm_block_acc(A, Bt, (KDIM), mbase, nbase, Bs, acc);

// ---------------- stage 0: x [B,C,H,W] -> t bf16 [B*N][C] ----------------
__global__ void __launch_bounds__(256) k_permute(const float* __restrict__ x,
                                                 __bf16* __restrict__ t) {
  size_t i = (size_t)blockIdx.x * 256 + threadIdx.x;
  if (i >= (size_t)BN * CCH) return;
  int col = (int)(i & 255);
  size_t row = i >> 8;
  int b = (int)(row / NTOK), n = (int)(row % NTOK);
  t[i] = (__bf16)x[((size_t)(b * CCH + col)) * NTOK + n];
}

// ---------------- stage 0b: W [K][N] fp32 -> Wt [N][K] bf16 ----------------
__global__ void __launch_bounds__(256) k_wt(const float* __restrict__ W,
                                            __bf16* __restrict__ Wt, int K, int N) {
  size_t i = (size_t)blockIdx.x * 256 + threadIdx.x;
  if (i >= (size_t)K * N) return;
  int k = (int)(i % K);
  int n = (int)(i / K);
  Wt[i] = (__bf16)W[(size_t)k * N + n];
}

// ---------------- stage 1: QKV GEMM + bias ----------------
__global__ void __launch_bounds__(256) k_gemm_qkv(const __bf16* __restrict__ A,
                                                  const __bf16* __restrict__ Bt,
                                                  const float* __restrict__ bias,
                                                  __bf16* __restrict__ out) {
  GEMM_PROLOGUE(CCH, 768)
#pragma unroll
  for (int mi = 0; mi < 2; ++mi)
#pragma unroll
    for (int t = 0; t < 4; ++t)
#pragma unroll
      for (int r = 0; r < 8; ++r) {
        int row = mbase + mi * 16 + hh * 8 + r;
        int col = nbase + t * 16 + rl;
        out[(size_t)row * 768 + col] = (__bf16)(acc[mi][t][r] + bias[col]);
      }
}

// ---------------- stage 2: attention, one wave per (b, head, 16-row q tile) ----
__global__ void __launch_bounds__(32) k_attn(const __bf16* __restrict__ qkv,
                                             __bf16* __restrict__ o) {
  __shared__ float  S[16 * 208];
  __shared__ __bf16 P[16 * 224];
  __shared__ __bf16 Vt[32 * 224];
  __shared__ float  rowsum[16];
  const int lane = threadIdx.x;
  const int mt = blockIdx.x % 13;
  const int h  = (blockIdx.x / 13) % HEADS;
  const int b  = blockIdx.x / (13 * HEADS);
  const int rl = lane & 15, hh = lane >> 4;
  v8f zero = {0.f, 0.f, 0.f, 0.f, 0.f, 0.f, 0.f, 0.f};

  for (int i = lane; i < 16 * 224; i += 32) P[i] = (__bf16)0.f;
  for (int i = lane; i < 32 * 224; i += 32) {
    int d = i / 224, kk = i % 224;
    Vt[i] = (kk < NTOK) ? qkv[(size_t)(b * NTOK + kk) * 768 + 512 + h * HD + d]
                        : (__bf16)0.f;
  }

  // S = Q * K^T : one WMMA (K=32) per 16-column tile
  Frag16 qf;
  {
    int m = mt * 16 + rl;
    if (m < NTOK) load_frag(qkv + (size_t)(b * NTOK + m) * 768 + h * HD, 0, hh, qf);
    else          zero_frag(qf);
  }
  for (int ntile = 0; ntile < 13; ++ntile) {
    Frag16 kf;
    int n = ntile * 16 + rl;
    if (n < NTOK) load_frag(qkv + (size_t)(b * NTOK + n) * 768 + 256 + h * HD, 0, hh, kf);
    else          zero_frag(kf);
    v8f s = zero;
    s = wmma_bf16(qf, kf, s);
#pragma unroll
    for (int r = 0; r < 8; ++r)
      S[(hh * 8 + r) * 208 + ntile * 16 + rl] = s[r];
  }
  __syncthreads();

  // softmax rows (unnormalized exp into P, sums kept for epilogue)
  if (lane < 16) {
    const float scale = 0.17677669529663687f;  // 1/sqrt(32)
    float mx = -3.0e38f;
    for (int j = 0; j < NTOK; ++j) mx = fmaxf(mx, S[lane * 208 + j]);
    float sum = 0.f;
    for (int j = 0; j < NTOK; ++j) {
      float e = __expf((S[lane * 208 + j] - mx) * scale);
      sum += e;
      P[lane * 224 + j] = (__bf16)e;
    }
    rowsum[lane] = sum;
  }
  __syncthreads();

  // O = P * V : K padded to 224, N=32 (two 16-wide tiles)
  v8f oa[2] = {zero, zero};
  const __bf16* Pr = &P[rl * 224];
  const __bf16* V0 = &Vt[rl * 224];
  const __bf16* V1 = &Vt[(16 + rl) * 224];
  for (int kb = 0; kb < 224; kb += 32) {
    Frag16 pa, vb0, vb1;
    load_frag(Pr, kb, hh, pa);
    load_frag(V0, kb, hh, vb0);
    load_frag(V1, kb, hh, vb1);
    oa[0] = wmma_bf16(pa, vb0, oa[0]);
    oa[1] = wmma_bf16(pa, vb1, oa[1]);
  }
#pragma unroll
  for (int t = 0; t < 2; ++t)
#pragma unroll
    for (int r = 0; r < 8; ++r) {
      int row = mt * 16 + hh * 8 + r;
      if (row < NTOK) {
        float inv = 1.f / rowsum[hh * 8 + r];
        o[(size_t)(b * NTOK + row) * CCH + h * HD + t * 16 + rl] = (__bf16)(oa[t][r] * inv);
      }
    }
}

// ---------------- stage 3: proj GEMM + residual + RepBN1 ----------------
__global__ void __launch_bounds__(256) k_gemm_proj(const __bf16* __restrict__ A,
                                                   const __bf16* __restrict__ Bt,
                                                   const float* __restrict__ bias,
                                                   const float* __restrict__ x,
                                                   const float* __restrict__ alpha1,
                                                   const float* __restrict__ gamma1,
                                                   const float* __restrict__ beta1,
                                                   const float* __restrict__ rm1,
                                                   const float* __restrict__ rv1,
                                                   float* __restrict__ t2f,
                                                   __bf16* __restrict__ t2b) {
  GEMM_PROLOGUE(CCH, CCH)
  const float a1 = alpha1[0];
#pragma unroll
  for (int mi = 0; mi < 2; ++mi)
#pragma unroll
    for (int t = 0; t < 4; ++t)
#pragma unroll
      for (int r = 0; r < 8; ++r) {
        int row = mbase + mi * 16 + hh * 8 + r;
        int col = nbase + t * 16 + rl;
        int b = row / NTOK, n = row % NTOK;
        float tval = x[((size_t)(b * CCH + col)) * NTOK + n];
        float s = tval + acc[mi][t][r] + bias[col];
        float g = gamma1[col] * rsqrtf(rv1[col] + EPSBN);
        float bn = (s - rm1[col]) * g + beta1[col] + a1 * s;
        t2f[(size_t)row * CCH + col] = bn;
        t2b[(size_t)row * CCH + col] = (__bf16)bn;
      }
}

// ---------------- stage 4: FFN1 GEMM + bias + SpatialSILU ----------------
__global__ void __launch_bounds__(256) k_gemm_ffn1(const __bf16* __restrict__ A,
                                                   const __bf16* __restrict__ Bt,
                                                   const float* __restrict__ bias,
                                                   const float* __restrict__ sa_w,
                                                   const float* __restrict__ sa_b,
                                                   __bf16* __restrict__ hbuf) {
  GEMM_PROLOGUE(CCH, CM)
#pragma unroll
  for (int mi = 0; mi < 2; ++mi)
#pragma unroll
    for (int t = 0; t < 4; ++t)
#pragma unroll
      for (int r = 0; r < 8; ++r) {
        int row = mbase + mi * 16 + hh * 8 + r;
        int col = nbase + t * 16 + rl;
        int b = row / NTOK;
        float hv = acc[mi][t][r] + bias[col];
        float wgt = sa_w[b] * hv + sa_b[b];
        float outv = hv / (1.f + __expf(-wgt * hv));
        hbuf[(size_t)row * CM + col] = (__bf16)outv;
      }
}

// ---------------- stage 5: FFN2 GEMM + residual + RepBN2 + un-permute ------
__global__ void __launch_bounds__(256) k_gemm_ffn2(const __bf16* __restrict__ A,
                                                   const __bf16* __restrict__ Bt,
                                                   const float* __restrict__ bias,
                                                   const float* __restrict__ t2f,
                                                   const float* __restrict__ alpha2,
                                                   const float* __restrict__ gamma2,
                                                   const float* __restrict__ beta2,
                                                   const float* __restrict__ rm2,
                                                   const float* __restrict__ rv2,
                                                   float* __restrict__ out) {
  GEMM_PROLOGUE(CM, CCH)
  const float a2 = alpha2[0];
#pragma unroll
  for (int mi = 0; mi < 2; ++mi)
#pragma unroll
    for (int t = 0; t < 4; ++t)
#pragma unroll
      for (int r = 0; r < 8; ++r) {
        int row = mbase + mi * 16 + hh * 8 + r;
        int col = nbase + t * 16 + rl;
        int b = row / NTOK, n = row % NTOK;
        float s = t2f[(size_t)row * CCH + col] + acc[mi][t][r] + bias[col];
        float g = gamma2[col] * rsqrtf(rv2[col] + EPSBN);
        float bn = (s - rm2[col]) * g + beta2[col] + a2 * s;
        out[((size_t)(b * CCH + col)) * NTOK + n] = bn;
      }
}

// ---------------- host launcher ----------------
extern "C" void kernel_launch(void* const* d_in, const int* in_sizes, int n_in,
                              void* d_out, int out_size, void* d_ws, size_t ws_size,
                              hipStream_t stream) {
  (void)in_sizes; (void)n_in; (void)out_size; (void)ws_size;
  const float* x      = (const float*)d_in[0];
  const float* W_qkv  = (const float*)d_in[1];
  const float* b_qkv  = (const float*)d_in[2];
  const float* W_proj = (const float*)d_in[3];
  const float* b_proj = (const float*)d_in[4];
  const float* W1     = (const float*)d_in[5];
  const float* b1     = (const float*)d_in[6];
  const float* W2     = (const float*)d_in[7];
  const float* b2     = (const float*)d_in[8];
  const float* sa_w   = (const float*)d_in[9];
  const float* sa_b   = (const float*)d_in[10];
  const float* alpha1 = (const float*)d_in[11];
  const float* gamma1 = (const float*)d_in[12];
  const float* beta1  = (const float*)d_in[13];
  const float* rm1    = (const float*)d_in[14];
  const float* rv1    = (const float*)d_in[15];
  const float* alpha2 = (const float*)d_in[16];
  const float* gamma2 = (const float*)d_in[17];
  const float* beta2  = (const float*)d_in[18];
  const float* rm2    = (const float*)d_in[19];
  const float* rv2    = (const float*)d_in[20];
  float* out = (float*)d_out;

  char* ws = (char*)d_ws;
  size_t off = 0;
  auto take = [&](size_t bytes) {
    char* p = ws + off;
    off = (off + bytes + 255) & ~(size_t)255;
    return p;
  };
  __bf16* tb     = (__bf16*)take((size_t)BN * CCH * 2);
  __bf16* qkvb   = (__bf16*)take((size_t)BN * 768 * 2);
  __bf16* ob     = (__bf16*)take((size_t)BN * CCH * 2);
  float*  t2f    = (float*) take((size_t)BN * CCH * 4);
  __bf16* t2b    = (__bf16*)take((size_t)BN * CCH * 2);
  __bf16* hb     = (__bf16*)take((size_t)BN * CM * 2);
  __bf16* wqkvt  = (__bf16*)take((size_t)768 * CCH * 2);
  __bf16* wprojt = (__bf16*)take((size_t)CCH * CCH * 2);
  __bf16* w1t    = (__bf16*)take((size_t)CM * CCH * 2);
  __bf16* w2t    = (__bf16*)take((size_t)CCH * CM * 2);

  // stage 0: layout conversions
  k_permute<<<(BN * CCH) / 256, 256, 0, stream>>>(x, tb);
  k_wt<<<(CCH * 768) / 256, 256, 0, stream>>>(W_qkv, wqkvt, CCH, 768);
  k_wt<<<(CCH * CCH) / 256, 256, 0, stream>>>(W_proj, wprojt, CCH, CCH);
  k_wt<<<(CCH * CM) / 256, 256, 0, stream>>>(W1, w1t, CCH, CM);
  k_wt<<<(CM * CCH) / 256, 256, 0, stream>>>(W2, w2t, CM, CCH);

  // stage 1: QKV   (196 row-blocks x 12 col-panels)
  k_gemm_qkv<<<196 * 12, 256, 0, stream>>>(tb, wqkvt, b_qkv, qkvb);

  // stage 2: attention (256 batches x 8 heads x 13 q-tiles, 1 wave each)
  k_attn<<<BSZ * HEADS * 13, 32, 0, stream>>>(qkvb, ob);

  // stage 3: proj + residual + RepBN1
  k_gemm_proj<<<196 * 4, 256, 0, stream>>>(ob, wprojt, b_proj, x,
                                           alpha1, gamma1, beta1, rm1, rv1,
                                           t2f, t2b);

  // stage 4: FFN1 + SpatialSILU
  k_gemm_ffn1<<<196 * 32, 256, 0, stream>>>(t2b, w1t, b1, sa_w, sa_b, hb);

  // stage 5: FFN2 + residual + RepBN2 + un-permute to [B,C,H,W]
  k_gemm_ffn2<<<196 * 4, 256, 0, stream>>>(hb, w2t, b2, t2f,
                                           alpha2, gamma2, beta2, rm2, rv2,
                                           out);
}
